// CausalSelfAttention_74801150427905
// MI455X (gfx1250) — matrix operand order repaired and checked by reading, not compiled, not run on hardware
//
#include <hip/hip_runtime.h>
#include <hip/hip_bf16.h>

// ---------------------------------------------------------------------------
// CDNA5 (gfx1250) fused causal self-attention, bf16 WMMA everywhere.
// B=2, T=2048, D=1024, H=16, DH=64.
// Round 3: double-buffered async-to-LDS GEMM pipeline; flash attention split
// into unmasked main loop + single masked tail tile.
// ---------------------------------------------------------------------------

typedef __attribute__((ext_vector_type(8)))  __bf16 bf16x8;
typedef __attribute__((ext_vector_type(16))) __bf16 bf16x16;
typedef __attribute__((ext_vector_type(8)))  float  f32x8;

static __device__ __forceinline__ bf16x8 load8(const __bf16* p) {
  return *reinterpret_cast<const bf16x8*>(p);
}
static __device__ __forceinline__ bf16x16 bfcat(bf16x8 lo, bf16x8 hi) {
  return __builtin_shufflevector(lo, hi, 0, 1, 2, 3, 4, 5, 6, 7,
                                         8, 9, 10, 11, 12, 13, 14, 15);
}
static __device__ __forceinline__ f32x8 wmma_bf16(bf16x16 a, bf16x16 b, f32x8 c) {
  // v_wmma_f32_16x16x32_bf16: D = A(16x32) * B(32x16) + C(16x16 f32)
  return __builtin_amdgcn_wmma_f32_16x16x32_bf16(
      /*neg_a=*/false, a, /*neg_b=*/false, b,
      /*c_mod=*/(short)0, c, /*reuse_a=*/false, /*reuse_b=*/false);
}
// Same-wave LDS producer->consumer: HW keeps DS ops in order (ISA 7.3); we only
// need the wait + a compiler reordering fence.
static __device__ __forceinline__ void wave_lds_fence() {
  asm volatile("s_wait_dscnt 0" ::: "memory");
}
// XOR-butterfly within 16-lane halves using ds_swizzle_b32 group-of-32 mode:
// imm = {0, xor[4:0], or[4:0]=0, and[4:0]=0x1f}  ->  lane_out = lane ^ xor
template <int IMM>
static __device__ __forceinline__ float swz_f32(float x) {
  return __builtin_bit_cast(
      float, __builtin_amdgcn_ds_swizzle(__builtin_bit_cast(int, x), IMM));
}
// Async global->LDS 16B copy (GLOBAL_LOAD_ASYNC_TO_LDS_B128, ASYNCcnt).
// lds_off = low 32 bits of the flat shared address (= LDS byte offset).
static __device__ __forceinline__ void async_b128(const __bf16* g,
                                                  unsigned lds_off) {
  asm volatile("global_load_async_to_lds_b128 %0, %1, off"
               :: "v"(lds_off), "v"(g)
               : "memory");
}
static __device__ __forceinline__ void wait_async0() {
  asm volatile("s_wait_asynccnt 0" ::: "memory");
}

// ------------------------------- constants ---------------------------------
constexpr int Bc = 2, Tc = 2048, Dc = 1024, Hc = 16, DHc = 64;

// ------------------------- f32 -> bf16 convert -----------------------------
__global__ __launch_bounds__(256) void cvt_f32_bf16_kernel(
    const float* __restrict__ in, __bf16* __restrict__ out, int n) {
  int i = blockIdx.x * blockDim.x + threadIdx.x;
  if (i < n) out[i] = (__bf16)in[i];
}

// ------------------------- tiled WMMA GEMM + bias --------------------------
// C[M,N](OutT) = A[M,K](bf16, row major) * B[K,N](bf16, row major) + bias[N]
// Block: 256 threads (8 waves). Tile: BM=128, BN=128, BK=32, double-buffered.
// A tile staged by async-to-LDS DMA; B tile staged transposed ([n][k]).
// Pipeline: issue tile k+1 (async A + global B loads) -> compute tile k
// (8 wmma) -> stage B(k+1) to LDS -> s_wait_asynccnt 0 -> barrier.
template <typename OutT>
__global__ __launch_bounds__(256) void gemm_bf16_bias_kernel(
    const __bf16* __restrict__ A, const __bf16* __restrict__ Bm,
    const float* __restrict__ bias, OutT* __restrict__ C,
    int M, int N, int K) {
  constexpr int BM = 128, BN = 128, BK = 32;
  __shared__ __align__(16) __bf16 ldsA[2][BM * BK];  // [m][k]
  __shared__ __align__(16) __bf16 ldsB[2][BN * BK];  // [n][k] (transposed)

  const int tid  = threadIdx.x;
  const int wid  = tid >> 5;
  const int lane = tid & 31;
  const int hf   = lane >> 4;   // K/M half selector per WMMA layouts
  const int r    = lane & 15;

  const int bm0 = blockIdx.y * BM;
  const int bn0 = blockIdx.x * BN;
  const int wm  = (wid >> 2) * 64;  // 0 | 64
  const int wn  = (wid & 3) * 32;   // 0..96

  // staging assignments (16 elements per thread per tile)
  const int arow = tid >> 1;         // 0..127
  const int acol = (tid & 1) * 16;   // 0 | 16
  const int bkr  = tid & 31;         // k row 0..31
  const int bnc  = (tid >> 5) * 16;  // n chunk 0..112

  // LDS byte offsets of this thread's A staging slot (flat addr low 32 bits)
  const unsigned ldsA_off0 = (unsigned)(size_t)(&ldsA[0][arow * BK + acol]);
  const unsigned ldsA_off1 = (unsigned)(size_t)(&ldsA[1][arow * BK + acol]);

  const __bf16* agbase = A + (size_t)(bm0 + arow) * K + acol;
  const __bf16* bgbase = Bm + (size_t)bkr * N + bn0 + bnc;

  f32x8 acc[4][2];
#pragma unroll
  for (int mt = 0; mt < 4; ++mt)
#pragma unroll
    for (int nt = 0; nt < 2; ++nt)
#pragma unroll
      for (int i = 0; i < 8; ++i) acc[mt][nt][i] = 0.0f;

  // ---- prologue: tile 0 into buffer 0 ----
  {
    async_b128(agbase, ldsA_off0);
    async_b128(agbase + 8, ldsA_off0 + 16);
    bf16x8 b0 = load8(bgbase), b1 = load8(bgbase + 8);
#pragma unroll
    for (int j = 0; j < 8; ++j) ldsB[0][(bnc + j) * BK + bkr] = b0[j];
#pragma unroll
    for (int j = 0; j < 8; ++j) ldsB[0][(bnc + 8 + j) * BK + bkr] = b1[j];
    wait_async0();
    __syncthreads();
  }

  const int nk = K / BK;
  for (int k = 0; k < nk; ++k) {
    const int cur = k & 1;
    const int nxt = cur ^ 1;
    const bool have_next = (k + 1 < nk);

    // --- issue next tile: async A DMA + B global loads (hide under wmma) ---
    bf16x8 nb0, nb1;
    if (have_next) {
      const __bf16* ag = agbase + (k + 1) * BK;
      async_b128(ag, nxt ? ldsA_off1 : ldsA_off0);
      async_b128(ag + 8, (nxt ? ldsA_off1 : ldsA_off0) + 16);
      const __bf16* bg = bgbase + (size_t)((k + 1) * BK) * N;
      nb0 = load8(bg);
      nb1 = load8(bg + 8);
    }

    // --- compute tile k from buffer cur ---
    // A 16x32 bf16: lane r -> row M=r; elem j -> K=16h+j
    bf16x16 afrag[4];
#pragma unroll
    for (int mt = 0; mt < 4; ++mt) {
      const __bf16* p = &ldsA[cur][(wm + mt * 16 + r) * BK + 16 * hf];
      afrag[mt] = bfcat(load8(p), load8(p + 8));
    }
    // B 32x16 bf16: lane = N + 16*(K half); elem j -> K=16h+j (contig in ldsB)
    bf16x16 bfrag[2];
#pragma unroll
    for (int nt = 0; nt < 2; ++nt) {
      const __bf16* p = &ldsB[cur][(wn + nt * 16 + r) * BK + 16 * hf];
      bfrag[nt] = bfcat(load8(p), load8(p + 8));
    }
#pragma unroll
    for (int mt = 0; mt < 4; ++mt)
#pragma unroll
      for (int nt = 0; nt < 2; ++nt)
        acc[mt][nt] = wmma_bf16(afrag[mt], bfrag[nt], acc[mt][nt]);

    // --- stage next B tile (transposed) into buffer nxt ---
    if (have_next) {
#pragma unroll
      for (int j = 0; j < 8; ++j) ldsB[nxt][(bnc + j) * BK + bkr] = nb0[j];
#pragma unroll
      for (int j = 0; j < 8; ++j) ldsB[nxt][(bnc + 8 + j) * BK + bkr] = nb1[j];
    }

    wait_async0();   // next A tile landed (had the whole compute phase)
    __syncthreads(); // publish next tile; all reads of cur buffer done
  }

  // --- epilogue: C layout VGPR i -> row = 8*hf + i, col = r ---
#pragma unroll
  for (int mt = 0; mt < 4; ++mt)
#pragma unroll
    for (int nt = 0; nt < 2; ++nt) {
      const int gn = bn0 + wn + nt * 16 + r;
      const float bv = bias[gn];
#pragma unroll
      for (int i = 0; i < 8; ++i) {
        const int gm = bm0 + wm + mt * 16 + hf * 8 + i;
        C[(size_t)gm * N + gn] = (OutT)(acc[mt][nt][i] + bv);
      }
    }
}

// -------------------- RoPE + head split / V transpose ----------------------
// qkv: [B,T,3D] bf16  ->  q,k: [B,H,T,DH] bf16 (RoPE applied),
//                         vt:  [B,H,DH,T] bf16 (transposed for flash B-frags)
__global__ __launch_bounds__(256) void rope_split_kernel(
    const __bf16* __restrict__ qkv, __bf16* __restrict__ qb,
    __bf16* __restrict__ kb, __bf16* __restrict__ vt) {
  int idx = blockIdx.x * blockDim.x + threadIdx.x;  // B*T*H*32 pairs
  if (idx >= Bc * Tc * Hc * (DHc / 2)) return;
  const int p = idx & 31;
  const int h = (idx >> 5) & (Hc - 1);
  const int t = (idx >> 9) & (Tc - 1);
  const int b = idx >> 20;
  const int d0 = 2 * p;

  const __bf16* base = qkv + ((size_t)(b * Tc + t)) * (3 * Dc) + h * DHc + d0;
  float q0 = (float)base[0],          q1 = (float)base[1];
  float k0 = (float)base[Dc],         k1 = (float)base[Dc + 1];
  float v0 = (float)base[2 * Dc],     v1 = (float)base[2 * Dc + 1];

  // inv_freq = base^{-d0/DH}; angle = t * inv_freq
  float inv_freq = __powf(10000.0f, -(float)d0 * (1.0f / (float)DHc));
  float ang = (float)t * inv_freq, s, c;
  __sincosf(ang, &s, &c);

  const size_t hb = (size_t)(b * Hc + h);
  const size_t qoff = (hb * Tc + t) * DHc + d0;
  qb[qoff]     = (__bf16)(q0 * c - q1 * s);
  qb[qoff + 1] = (__bf16)(q1 * c + q0 * s);
  kb[qoff]     = (__bf16)(k0 * c - k1 * s);
  kb[qoff + 1] = (__bf16)(k1 * c + k0 * s);

  const size_t voff = (hb * DHc + d0) * Tc + t;
  vt[voff]      = (__bf16)v0;
  vt[voff + Tc] = (__bf16)v1;
}

// --------------------------- flash attention -------------------------------
// Grid: (T/64, B*H). Block: 128 threads = 4 waves; each wave owns 16 q rows.
// kv tiles of 32 keys. With 16 q rows per wave, exactly ONE tile per wave is
// causally partial (kv0 = q0 & ~31): main loop runs unmasked (no cmp/cndmask),
// masked epilogue runs once. Per tile: S = Q*K^T (4 wmma), ds_swizzle max
// butterfly, exp, P->LDS (C->A layout), row sums via P x ones (1 wmma),
// O += P*V (4 wmma).
__global__ __launch_bounds__(128) void flash_attn_kernel(
    const __bf16* __restrict__ qg,  // [B,H,T,DH]
    const __bf16* __restrict__ kg,  // [B,H,T,DH]
    const __bf16* __restrict__ vt,  // [B,H,DH,T]
    __bf16* __restrict__ out) {     // [B,T,D]
  __shared__ __align__(16) __bf16 ldsP[4][16 * 32];

  const int wid  = threadIdx.x >> 5;
  const int lane = threadIdx.x & 31;
  const int hf   = lane >> 4;
  const int r    = lane & 15;

  const int bh = blockIdx.y;
  const int b = bh >> 4, h = bh & (Hc - 1);
  const int q0 = blockIdx.x * 64 + wid * 16;  // this wave's first q row

  const __bf16* qbase = qg + ((size_t)bh * Tc) * DHc;
  const __bf16* kbase = kg + ((size_t)bh * Tc) * DHc;
  const __bf16* vbase = vt + ((size_t)bh * DHc) * Tc;

  // Q A-fragments (lane r -> row q0+r), two K-dim steps of 32
  bf16x16 qa[2];
  {
    const __bf16* p = qbase + (size_t)(q0 + r) * DHc;
    qa[0] = bfcat(load8(p + 16 * hf),      load8(p + 16 * hf + 8));
    qa[1] = bfcat(load8(p + 32 + 16 * hf), load8(p + 32 + 16 * hf + 8));
  }

  // all-ones B fragment for row sums
  bf16x16 onesf;
#pragma unroll
  for (int j = 0; j < 16; ++j) onesf[j] = (__bf16)1.0f;

  f32x8 o[4];
  float mrun[8], lrun[8], alph[8];
#pragma unroll
  for (int t = 0; t < 4; ++t)
#pragma unroll
    for (int i = 0; i < 8; ++i) o[t][i] = 0.0f;
#pragma unroll
  for (int i = 0; i < 8; ++i) { mrun[i] = -1e30f; lrun[i] = 0.0f; }

  const float scale = 0.125f;  // 1/sqrt(64)

  // Processes one 32-key kv tile; `masked` is a compile-time constant at every
  // call site after inlining, so the main loop carries no mask VALU.
  auto kv_tile = [&](int kv0, bool masked) {
    // ---- S = Q * K^T for 32 keys (two 16-col n-tiles) ----
    f32x8 sacc[2];
#pragma unroll
    for (int nt = 0; nt < 2; ++nt) {
#pragma unroll
      for (int i = 0; i < 8; ++i) sacc[nt][i] = 0.0f;
      // B-frag: elem j -> K_mat[key = kv0+nt*16+r][d = 16h+j (+32 step)]
      const __bf16* kp = kbase + (size_t)(kv0 + nt * 16 + r) * DHc;
      bf16x16 kb0 = bfcat(load8(kp + 16 * hf),      load8(kp + 16 * hf + 8));
      bf16x16 kb1 = bfcat(load8(kp + 32 + 16 * hf), load8(kp + 32 + 16 * hf + 8));
      sacc[nt] = wmma_bf16(qa[0], kb0, sacc[nt]);
      sacc[nt] = wmma_bf16(qa[1], kb1, sacc[nt]);
    }

    // ---- online softmax: (mask), row max via swizzle butterfly, exp ----
#pragma unroll
    for (int i = 0; i < 8; ++i) {
      float v0 = sacc[0][i] * scale;
      float v1 = sacc[1][i] * scale;
      if (masked) {
        const int rowg = q0 + hf * 8 + i;
        if (kv0 + r > rowg)      v0 = -1e30f;
        if (kv0 + 16 + r > rowg) v1 = -1e30f;
      }
      float mx = fmaxf(v0, v1);
      mx = fmaxf(mx, swz_f32<0x041F>(mx));  // lane ^ 1
      mx = fmaxf(mx, swz_f32<0x081F>(mx));  // lane ^ 2
      mx = fmaxf(mx, swz_f32<0x101F>(mx));  // lane ^ 4
      mx = fmaxf(mx, swz_f32<0x201F>(mx));  // lane ^ 8
      const float mnew = fmaxf(mrun[i], mx);
      alph[i] = __expf(mrun[i] - mnew);
      mrun[i] = mnew;
      const float p0 = __expf(v0 - mnew);
      const float p1 = __expf(v1 - mnew);
      o[0][i] *= alph[i]; o[1][i] *= alph[i];
      o[2][i] *= alph[i]; o[3][i] *= alph[i];
      // P tile to LDS: row = hf*8+i, key col = r / 16+r
      ldsP[wid][(hf * 8 + i) * 32 + r]      = (__bf16)p0;
      ldsP[wid][(hf * 8 + i) * 32 + 16 + r] = (__bf16)p1;
    }

    wave_lds_fence();  // same-wave DS in-order; wait stores, stop reordering

    // P as A-fragment (16x32): lane r -> row r; elem j -> kk=16h+j
    const __bf16* pp = &ldsP[wid][r * 32 + 16 * hf];
    bf16x16 pa = bfcat(load8(pp), load8(pp + 8));
    asm volatile("" ::: "memory");  // keep next iter's stores after this load

    // ---- row sums of P via WMMA with ones: C-layout replication gives every
    //      lane of a half the sum for row (i + 8*hf) in element i ----
    f32x8 rz;
#pragma unroll
    for (int i = 0; i < 8; ++i) rz[i] = 0.0f;
    f32x8 rs = wmma_bf16(pa, onesf, rz);
#pragma unroll
    for (int i = 0; i < 8; ++i) lrun[i] = lrun[i] * alph[i] + rs[i];

    // ---- O += P * V (4 dh tiles of 16) ----
#pragma unroll
    for (int t = 0; t < 4; ++t) {
      // B-frag: elem j -> V[key = kv0+16h+j][d = t*16+r]; vt is [d][t] contig
      const __bf16* vp = vbase + (size_t)(t * 16 + r) * Tc + kv0 + 16 * hf;
      bf16x16 vb = bfcat(load8(vp), load8(vp + 8));
      o[t] = wmma_bf16(pa, vb, o[t]);
    }
  };

  // main loop: every key in [kv0, kv0+31] is <= every row (kv0+31 <= q0)
  const int kvfull = q0 & ~31;
  for (int kv0 = 0; kv0 < kvfull; kv0 += 32) kv_tile(kv0, false);
  // single causally-partial tail tile
  kv_tile(kvfull, true);

  // ---- normalize + write [B,T,D] bf16 ----
#pragma unroll
  for (int i = 0; i < 8; ++i) {
    const float inv = 1.0f / lrun[i];
    const int rowg = q0 + hf * 8 + i;
    const size_t base = ((size_t)b * Tc + rowg) * Dc + h * DHc + r;
    out[base]      = (__bf16)(o[0][i] * inv);
    out[base + 16] = (__bf16)(o[1][i] * inv);
    out[base + 32] = (__bf16)(o[2][i] * inv);
    out[base + 48] = (__bf16)(o[3][i] * inv);
  }
}

// ------------------------------- launcher ----------------------------------
extern "C" void kernel_launch(void* const* d_in, const int* in_sizes, int n_in,
                              void* d_out, int out_size, void* d_ws,
                              size_t ws_size, hipStream_t stream) {
  (void)in_sizes; (void)n_in; (void)out_size; (void)ws_size;
  const float* x     = (const float*)d_in[0];  // [B,T,D]
  const float* Wqkv  = (const float*)d_in[1];  // [D,3D]
  const float* bqkv  = (const float*)d_in[2];  // [3D]
  const float* Wproj = (const float*)d_in[3];  // [D,D]
  const float* bproj = (const float*)d_in[4];  // [D]
  float* outp = (float*)d_out;                 // [B,T,D]

  const size_t BT = (size_t)Bc * Tc;  // 4096
  char* ws = (char*)d_ws;
  size_t off = 0;
  auto take = [&](size_t bytes) -> char* {
    char* p = ws + off;
    off += (bytes + 255) & ~(size_t)255;
    return p;
  };
  __bf16* xb     = (__bf16*)take(BT * Dc * 2);
  __bf16* wqkvb  = (__bf16*)take((size_t)Dc * 3 * Dc * 2);
  __bf16* wprojb = (__bf16*)take((size_t)Dc * Dc * 2);
  __bf16* qkvb   = (__bf16*)take(BT * 3 * Dc * 2);
  __bf16* qb     = (__bf16*)take(BT * Dc * 2);
  __bf16* kb     = (__bf16*)take(BT * Dc * 2);
  __bf16* vtb    = (__bf16*)take(BT * Dc * 2);
  __bf16* attb   = (__bf16*)take(BT * Dc * 2);

  // 1) convert inputs to bf16
  cvt_f32_bf16_kernel<<<(int)((BT * Dc + 255) / 256), 256, 0, stream>>>(
      x, xb, (int)(BT * Dc));
  cvt_f32_bf16_kernel<<<(Dc * 3 * Dc + 255) / 256, 256, 0, stream>>>(
      Wqkv, wqkvb, Dc * 3 * Dc);
  cvt_f32_bf16_kernel<<<(Dc * Dc + 255) / 256, 256, 0, stream>>>(
      Wproj, wprojb, Dc * Dc);

  // 2) QKV = x @ Wqkv + bqkv   (M=4096, N=3072, K=1024) -> bf16
  gemm_bf16_bias_kernel<__bf16><<<dim3(3 * Dc / 128, (int)(BT / 128)), 256, 0,
                                  stream>>>(xb, wqkvb, bqkv, qkvb,
                                            (int)BT, 3 * Dc, Dc);

  // 3) RoPE + split heads (+ V transpose)
  rope_split_kernel<<<(Bc * Tc * Hc * (DHc / 2)) / 256, 256, 0, stream>>>(
      qkvb, qb, kb, vtb);

  // 4) causal flash attention -> attb [B,T,D] bf16
  flash_attn_kernel<<<dim3(Tc / 64, Bc * Hc), 128, 0, stream>>>(qb, kb, vtb,
                                                                attb);

  // 5) out = attb @ Wproj + bproj  (M=4096, N=1024, K=1024) -> f32
  gemm_bf16_bias_kernel<float><<<dim3(Dc / 128, (int)(BT / 128)), 256, 0,
                                 stream>>>(attb, wprojb, bproj, outp,
                                           (int)BT, Dc, Dc);
}
